// CategorySpecificMLP_57982058496591
// MI455X (gfx1250) — matrix operand order repaired and checked by reading, not compiled, not run on hardware
//
#include <hip/hip_runtime.h>
#include <hip/hip_bf16.h>

typedef __attribute__((ext_vector_type(16))) _Float16 v16h;
typedef __attribute__((ext_vector_type(8)))  _Float16 v8h;
typedef __attribute__((ext_vector_type(4)))  _Float16 v4h;
typedef __attribute__((ext_vector_type(8)))  float    v8f;
typedef __attribute__((ext_vector_type(4)))  float    v4f;

#define DIN    1024
#define DOUT   1024
#define TOKENS 256      // tokens per sample (T)
#define NSAMP  64       // samples (B)
#define TM     128      // workgroup tile M
#define TN     128      // workgroup tile N
#define TK     32       // K step (f16 WMMA K)
#define NK     (DIN / TK)
#define LDA    40       // padded halfword stride for LDS A ([M][K])
#define LDB    40       // padded halfword stride for LDS B ([N][K])

__device__ __forceinline__ v16h cat8(v8h lo, v8h hi) {
    v16h r;
#pragma unroll
    for (int i = 0; i < 8; ++i) { r[i] = lo[i]; r[i + 8] = hi[i]; }
    return r;
}

// Global f32 tile fetch into registers (float4 granularity, coalesced).
__device__ __forceinline__ void load_tile(const float* __restrict__ Ab,
                                          const float* __restrict__ Wb,
                                          int k0, int tid, v4f ra[4], v4f rb[4]) {
#pragma unroll
    for (int j = 0; j < 4; ++j) {
        int idx = tid + 256 * j;           // 0..1023 float4 slots
        int r   = idx >> 3;                // A tile: 128 rows x 8 float4
        int c4  = idx & 7;
        ra[j] = *(const v4f*)(Ab + (size_t)r * DIN + k0 + c4 * 4);
        int kb  = idx >> 5;                // W tile: 32 rows x 32 float4
        int cb4 = idx & 31;
        rb[j] = *(const v4f*)(Wb + (size_t)(k0 + kb) * DOUT + cb4 * 4);
    }
}

// Convert f32 -> f16 (packed cvt) and store to LDS; W tile transposed to [N][K].
__device__ __forceinline__ void store_tile(_Float16* sA, _Float16* sB,
                                           int tid, const v4f ra[4], const v4f rb[4]) {
#pragma unroll
    for (int j = 0; j < 4; ++j) {
        int idx = tid + 256 * j;
        int r   = idx >> 3;
        int c4  = idx & 7;
        auto a01 = __builtin_amdgcn_cvt_pkrtz(ra[j][0], ra[j][1]);
        auto a23 = __builtin_amdgcn_cvt_pkrtz(ra[j][2], ra[j][3]);
        v4h ha;
        ha[0] = a01[0]; ha[1] = a01[1]; ha[2] = a23[0]; ha[3] = a23[1];
        *(v4h*)&sA[r * LDA + c4 * 4] = ha;          // 8B-aligned (80*r + 8*c4)

        int kb  = idx >> 5;
        int cb4 = idx & 31;
        auto b01 = __builtin_amdgcn_cvt_pkrtz(rb[j][0], rb[j][1]);
        auto b23 = __builtin_amdgcn_cvt_pkrtz(rb[j][2], rb[j][3]);
        sB[(cb4 * 4 + 0) * LDB + kb] = b01[0];
        sB[(cb4 * 4 + 1) * LDB + kb] = b01[1];
        sB[(cb4 * 4 + 2) * LDB + kb] = b23[0];
        sB[(cb4 * 4 + 3) * LDB + kb] = b23[1];
    }
}

template <int SILU>
__launch_bounds__(256)
__global__ void cat_mlp_layer(const float* __restrict__ hin,   // [B, T, DIN] f32
                              const float* __restrict__ W,     // [C, DIN, DOUT] f32
                              const float* __restrict__ bias,  // [C, DOUT] f32
                              const int*   __restrict__ cat_ids,
                              float* __restrict__ hout)        // [B, T, DOUT] f32
{
    // double-buffered tiles: 2 * (10240 + 10240) bytes = 40 KB
    __shared__ __align__(16) _Float16 sA[2][TM * LDA];
    __shared__ __align__(16) _Float16 sB[2][TN * LDB];

    const int b   = blockIdx.z;
    const int cat = cat_ids[b];
    const int m0  = blockIdx.y * TM;
    const int n0  = blockIdx.x * TN;

    const float* Ab    = hin  + ((size_t)b * TOKENS + m0) * DIN;
    const float* Wb    = W    + (size_t)cat * DIN * DOUT + n0;
    const float* biasb = bias + (size_t)cat * DOUT + n0;
    float*       Ob    = hout + ((size_t)b * TOKENS + m0) * DOUT + n0;

    const int tid  = threadIdx.x;
    const int lane = tid & 31;
    const int wid  = tid >> 5;
    const int wm   = wid & 3;   // 4 waves along M: rows wm*32 .. +31
    const int wn   = wid >> 2;  // 2 waves along N: cols wn*64 .. +63
    const int half = lane >> 4;
    const int l16  = lane & 15;

    v8f acc[2][4] = {};

    v4f ra[4], rb[4];
    load_tile(Ab, Wb, 0, tid, ra, rb);
    store_tile(sA[0], sB[0], tid, ra, rb);

    for (int kk = 0; kk < NK; ++kk) {
        const int cur = kk & 1;
        if (kk + 1 < NK)                       // fetch next tile early (hides latency)
            load_tile(Ab, Wb, (kk + 1) * TK, tid, ra, rb);

        __syncthreads();                       // buffer `cur` fully written

        // ---- fragments from LDS ----
        // A 16x32 f16: lanes 0-15 hold K{0..7,16..23}, lanes 16-31 hold K{8..15,24..31}
        const _Float16* cA = sA[cur];
        const _Float16* cB = sB[cur];
        v16h afrag[2];
#pragma unroll
        for (int tm = 0; tm < 2; ++tm) {
            int row = wm * 32 + tm * 16 + l16;
            v8h lo = *(const v8h*)&cA[row * LDA + half * 8];
            v8h hi = *(const v8h*)&cA[row * LDA + 16 + half * 8];
            afrag[tm] = cat8(lo, hi);
        }
        // B 32x16 f16: lanes 0-15 hold K{0..7,16..23}, lanes 16-31 hold K{8..15,24..31} (col = lane%16)
        v16h bfrag[4];
#pragma unroll
        for (int tn = 0; tn < 4; ++tn) {
            int col = wn * 64 + tn * 16 + l16;
            v8h lo = *(const v8h*)&cB[col * LDB + half * 8];
            v8h hi = *(const v8h*)&cB[col * LDB + 16 + half * 8];
            bfrag[tn] = cat8(lo, hi);
        }
#pragma unroll
        for (int tm = 0; tm < 2; ++tm)
#pragma unroll
            for (int tn = 0; tn < 4; ++tn)
                acc[tm][tn] = __builtin_amdgcn_wmma_f32_16x16x32_f16(
                    false, afrag[tm], false, bfrag[tn],
                    (short)0, acc[tm][tn], false, false);

        if (kk + 1 < NK)                       // stage next tile into the idle buffer
            store_tile(sA[cur ^ 1], sB[cur ^ 1], tid, ra, rb);
    }

    // ---- epilogue: bias + SiLU (fast sigmoid via v_rcp_f32), f32 store ----
    // C/D layout: VGPR r -> row (half*8 + r), column = l16 within each 16x16 tile
#pragma unroll
    for (int tn = 0; tn < 4; ++tn) {
        int col = wn * 64 + tn * 16 + l16;
        float bv = biasb[col];
#pragma unroll
        for (int tm = 0; tm < 2; ++tm) {
#pragma unroll
            for (int r = 0; r < 8; ++r) {
                int row = wm * 32 + tm * 16 + half * 8 + r;
                float v = acc[tm][tn][r] + bv;
                if (SILU)
                    v = v * __builtin_amdgcn_rcpf(1.0f + __expf(-v));
                Ob[(size_t)row * DOUT + col] = v;
            }
        }
    }
}

extern "C" void kernel_launch(void* const* d_in, const int* in_sizes, int n_in,
                              void* d_out, int out_size, void* d_ws, size_t ws_size,
                              hipStream_t stream) {
    const float* x   = (const float*)d_in[0];
    const int*   cat = (const int*)  d_in[1];
    const float* W1  = (const float*)d_in[2];
    const float* b1  = (const float*)d_in[3];
    const float* W2  = (const float*)d_in[4];
    const float* b2  = (const float*)d_in[5];
    const float* W3  = (const float*)d_in[6];
    const float* b3  = (const float*)d_in[7];
    const float* W4  = (const float*)d_in[8];
    const float* b4  = (const float*)d_in[9];
    float* out = (float*)d_out;

    // ping-pong intermediates in workspace: 2 x (64*256*1024) f32 = 128 MB
    const size_t hsz = (size_t)NSAMP * TOKENS * DOUT;
    float* h0 = (float*)d_ws;
    float* h1 = h0 + hsz;

    dim3 grid(DOUT / TN, TOKENS / TM, NSAMP);  // (8, 2, 64)
    dim3 blk(256);

    cat_mlp_layer<1><<<grid, blk, 0, stream>>>(x,  W1, b1, cat, h0);
    cat_mlp_layer<1><<<grid, blk, 0, stream>>>(h0, W2, b2, cat, h1);
    cat_mlp_layer<1><<<grid, blk, 0, stream>>>(h1, W3, b3, cat, h0);
    cat_mlp_layer<0><<<grid, blk, 0, stream>>>(h0, W4, b4, cat, out);
}
